// HADARDecomposer_47175920779638
// MI455X (gfx1250) — matrix operand order repaired and checked by reading, not compiled, not run on hardware
//
#include <hip/hip_runtime.h>
#include <hip/hip_bf16.h>

typedef __attribute__((ext_vector_type(16))) _Float16 v16h;
typedef __attribute__((ext_vector_type(8)))  float    v8f;

#define PLANCK_C1 1.191042e-8f
#define PLANCK_C2 1.4387752f
#define NCH   54
#define NMAT  6
#define NTMP  20
#define NCAND 120   // NMAT*NTMP
#define KPAD  64    // K padded for 2x WMMA 16x16x32
#define JPAD  128   // candidates padded to 8 tiles of 16
#define PIX_PER_BLOCK 64

// ---------------------------------------------------------------------------
// Kernel 1: build candidate spectra, their squared norms, emissivity means,
// and the WMMA-swizzled f16 B operand (candB).
// candB index: ((slice*2 + kstep)*32 + lane)*16 + h
//   N (cand col) = slice*16 + lane%16
//   K            = kstep*32 + ((lane&16)?16:0) + h      (contiguous per lane)
// ---------------------------------------------------------------------------
__global__ __launch_bounds__(256) void hadar_precompute(
    const float* __restrict__ s_sky, const float* __restrict__ s_ground,
    const float* __restrict__ library, const float* __restrict__ wg,
    float* __restrict__ cand, float* __restrict__ candsq,
    float* __restrict__ emean, _Float16* __restrict__ candB)
{
    const int tid = threadIdx.x;

    // Phase 1: cand[j][c] = e*B(nu,T) + (1-e)*x_amb
    for (int i = tid; i < NCAND * NCH; i += 256) {
        int j = i / NCH, c = i % NCH;
        int m = j / NTMP, t = j % NTMP;
        float T  = 250.0f + (100.0f / 19.0f) * (float)t;
        float nu = wg[c];
        float Bp = PLANCK_C1 * nu * nu * nu / expm1f(PLANCK_C2 * nu / T);
        float e  = library[m * NCH + c];
        float xa = 0.5f * (s_sky[c] + s_ground[c]);
        cand[i] = e * Bp + (1.0f - e) * xa;
    }
    __syncthreads();

    // Phase 2: candsq[j], emean[m]
    if (tid < NCAND) {
        float s = 0.0f;
        for (int c = 0; c < NCH; ++c) { float v = cand[tid * NCH + c]; s += v * v; }
        candsq[tid] = s;
    }
    if (tid >= NCAND && tid < NCAND + NMAT) {
        int m = tid - NCAND;
        float s = 0.0f;
        for (int c = 0; c < NCH; ++c) s += library[m * NCH + c];
        emean[m] = s / (float)NCH;
    }

    // Phase 3: swizzled + zero-padded f16 B operand
    for (int i = tid; i < 8 * 2 * 32 * 16; i += 256) {
        int h    = i & 15;
        int lane = (i >> 4) & 31;
        int k    = (i >> 9) & 1;
        int s    = i >> 10;
        int Nc = s * 16 + (lane & 15);
        int K  = k * 32 + ((lane & 16) ? 16 : 0) + h;
        float v = (Nc < NCAND && K < NCH) ? cand[Nc * NCH + K] : 0.0f;
        candB[i] = (_Float16)v;
    }
}

// ---------------------------------------------------------------------------
// Kernel 2: per 64 pixels: f16 WMMA GEMM vs all 128 (padded) candidates,
// argmin, and all per-pixel outputs. 256 threads = 8 wave32, one candidate
// slice (16 cols) per wave, looped over 4 pixel tiles of 16.
// ---------------------------------------------------------------------------
__global__ __launch_bounds__(256) void hadar_decompose(
    const float* __restrict__ s_obs, const float* __restrict__ library,
    const float* __restrict__ cand, const float* __restrict__ candsq,
    const float* __restrict__ emean, const _Float16* __restrict__ candB,
    float* __restrict__ out_t, float* __restrict__ out_e,
    float* __restrict__ out_v, float* __restrict__ out_beta,
    float* __restrict__ out_tex, float* __restrict__ out_recon,
    float* __restrict__ partials, long long Npix)
{
    __shared__ float tileA[PIX_PER_BLOCK * KPAD];   // 16 KB (zero-padded K)
    __shared__ float lossT[PIX_PER_BLOCK * JPAD];   // 32 KB
    __shared__ int   bestIdx[PIX_PER_BLOCK];
    __shared__ float bestLoss[PIX_PER_BLOCK];

    const int tid = threadIdx.x;
    const long long pixBase = (long long)blockIdx.x * PIX_PER_BLOCK;

    // Stage s_obs tile into LDS (f32, K padded to 64 with zeros).
    for (int i = tid; i < PIX_PER_BLOCK * KPAD; i += 256) {
        int p = i >> 6, c = i & 63;
        long long n = pixBase + p;
        tileA[i] = (c < NCH && n < Npix) ? s_obs[n * NCH + c] : 0.0f;
    }
    __syncthreads();

    const int lane  = tid & 31;
    const int slice = tid >> 5;           // wave id = candidate slice 0..7
    const int mrow  = lane & 15;
    const int kb    = (lane & 16) ? 16 : 0;   // A-fragment half-wave K offset
    const int mbase = (lane & 16) ? 8 : 0;    // D-fragment row offset
    const int j     = slice * 16 + (lane & 15);
    const float csq = (j < NCAND) ? candsq[j] : 0.0f;

    // Wave-invariant B fragments (two K-steps), loaded once.
    const v16h* bp = (const v16h*)candB;
    const v16h b0 = bp[(slice * 2 + 0) * 32 + lane];
    const v16h b1 = bp[(slice * 2 + 1) * 32 + lane];

    for (int pt = 0; pt < PIX_PER_BLOCK / 16; ++pt) {
        const float* arow = &tileA[(pt * 16 + mrow) * KPAD];
        v16h a0, a1;
#pragma unroll
        for (int h = 0; h < 8; ++h) {
            a0[h]     = (_Float16)arow[kb + h];          // K = kb..kb+7
            a0[h + 8] = (_Float16)arow[8 + kb + h];      // K = 8+kb..
            a1[h]     = (_Float16)arow[32 + kb + h];     // K = 32+kb..
            a1[h + 8] = (_Float16)arow[40 + kb + h];     // K = 40+kb..
        }
        v8f acc = {};
        acc = __builtin_amdgcn_wmma_f32_16x16x32_f16(false, a0, false, b0,
                                                     (short)0, acc, false, false);
        acc = __builtin_amdgcn_wmma_f32_16x16x32_f16(false, a1, false, b1,
                                                     (short)0, acc, false, false);
#pragma unroll
        for (int r = 0; r < 8; ++r)
            lossT[(pt * 16 + mbase + r) * JPAD + j] = csq - 2.0f * acc[r];
    }
    __syncthreads();

    // First-min argmin per pixel (strict '<' keeps first occurrence, matching
    // jnp.argmin over row-major m*T+t candidate order).
    if (tid < PIX_PER_BLOCK) {
        long long n = pixBase + tid;
        const float* lrow = &lossT[tid * JPAD];
        float best = lrow[0];
        int bi = 0;
        for (int jj = 1; jj < NCAND; ++jj) {
            float v = lrow[jj];
            if (v < best) { best = v; bi = jj; }
        }
        float ss = 0.0f;
        const float* arow = &tileA[tid * KPAD];
        for (int c = 0; c < NCH; ++c) ss += arow[c] * arow[c];
        bestIdx[tid]  = bi;
        bestLoss[tid] = (n < Npix) ? (best + ss) : 0.0f;
        if (n < Npix) {
            int t = bi % NTMP, m = bi / NTMP;
            out_t[n]    = 250.0f + (100.0f / 19.0f) * (float)t;
            out_v[n]    = 0.5f;
            out_beta[n] = 0.0f;
            out_tex[n]  = emean[m];
        }
    }
    __syncthreads();

    if (tid == 0) {
        float s = 0.0f;
        for (int p = 0; p < PIX_PER_BLOCK; ++p) s += bestLoss[p];
        partials[blockIdx.x] = s;
    }

    // best_e = library row of winner; s_recon = winning candidate row exactly.
    for (int i = tid; i < PIX_PER_BLOCK * NCH; i += 256) {
        int p = i / NCH, c = i % NCH;
        long long n = pixBase + p;
        if (n < Npix) {
            int bi = bestIdx[p];
            int m  = bi / NTMP;
            out_e[n * NCH + c]     = library[m * NCH + c];
            out_recon[n * NCH + c] = cand[bi * NCH + c];
        }
    }
}

// ---------------------------------------------------------------------------
// Kernel 3: deterministic fixed-order reduction of per-block loss sums.
// ---------------------------------------------------------------------------
__global__ __launch_bounds__(256) void hadar_reduce_obj(
    const float* __restrict__ partials, int nblocks, float invN,
    float* __restrict__ out_obj)
{
    __shared__ float sm[256];
    int tid = threadIdx.x;
    float s = 0.0f;
    for (int i = tid; i < nblocks; i += 256) s += partials[i];
    sm[tid] = s;
    __syncthreads();
    for (int off = 128; off > 0; off >>= 1) {
        if (tid < off) sm[tid] += sm[tid + off];
        __syncthreads();
    }
    if (tid == 0) out_obj[0] = sm[0] * invN;
}

extern "C" void kernel_launch(void* const* d_in, const int* in_sizes, int n_in,
                              void* d_out, int out_size, void* d_ws, size_t ws_size,
                              hipStream_t stream) {
    (void)n_in; (void)out_size; (void)ws_size;
    const float* s_obs    = (const float*)d_in[0];
    const float* s_sky    = (const float*)d_in[1];
    const float* s_ground = (const float*)d_in[2];
    const float* library  = (const float*)d_in[3];
    const float* wg       = (const float*)d_in[4];

    const long long N = (long long)in_sizes[0] / NCH;

    // Output layout (concatenated flat, reference return order):
    float* out      = (float*)d_out;
    float* out_t    = out;                         // [N]
    float* out_e    = out + N;                     // [N,54]
    float* out_v    = out + 55 * N;                // [N]
    float* out_beta = out + 56 * N;                // [N,1]
    float* out_tex  = out + 57 * N;                // [N]
    float* out_rec  = out + 58 * N;                // [N,54]
    float* out_obj  = out + 112 * N;               // scalar

    // Workspace layout (bytes): cand 25920 | candsq 480 | emean 24 |
    // pad | candB(f16) @26624 (16384) | partials @43008
    float*    cand     = (float*)d_ws;
    float*    candsq   = cand + NCAND * NCH;
    float*    emean    = cand + NCAND * NCH + NCAND;
    _Float16* candB    = (_Float16*)((char*)d_ws + 26624);
    float*    partials = (float*)((char*)d_ws + 43008);

    const int nblocks = (int)((N + PIX_PER_BLOCK - 1) / PIX_PER_BLOCK);

    hadar_precompute<<<1, 256, 0, stream>>>(s_sky, s_ground, library, wg,
                                            cand, candsq, emean, candB);
    hadar_decompose<<<nblocks, 256, 0, stream>>>(s_obs, library, cand, candsq,
                                                 emean, candB, out_t, out_e,
                                                 out_v, out_beta, out_tex,
                                                 out_rec, partials, N);
    hadar_reduce_obj<<<1, 256, 0, stream>>>(partials, nblocks,
                                            1.0f / (float)N, out_obj);
}
